// LoRAGenerator_78924319031461
// MI455X (gfx1250) — matrix operand order repaired and checked by reading, not compiled, not run on hardware
//
#include <hip/hip_runtime.h>

typedef __attribute__((ext_vector_type(16))) _Float16 v16h;
typedef __attribute__((ext_vector_type(8)))  float    v8f;
typedef __attribute__((ext_vector_type(4)))  float    f4v;

#define NEGV (-1000000000.0f)

__device__ __forceinline__ float gelu_exact(float x) {
    return 0.5f * x * (1.0f + erff(x * 0.70710678118654752f));
}

// ---------------------------------------------------------------------------
// Generic WMMA GEMM: C[M,N] = act(A[M,K] @ W[N,K]^T + bias[N] (+ res[M,N]))
// One wave (32 lanes) computes one 16x16 tile of C, K stepped by 32 using
// v_wmma_f32_16x16x32_f16. A/W are f32 in memory, converted to f16 in-register.
// ACT: 0 = none, 1 = relu, 2 = gelu(exact)
// Epilogue: uniform full-tile fast path (no exec churn), pointer-marched rows.
// ---------------------------------------------------------------------------
template <int ACT, bool RES>
__global__ void k_gemm(const float* __restrict__ A, const float* __restrict__ W,
                       const float* __restrict__ bias, const float* __restrict__ res,
                       float* __restrict__ C, int M, int N, int K) {
    const int lane = threadIdx.x;          // 0..31
    const int half = lane >> 4;            // 0/1
    const int l16  = lane & 15;
    const int tn = blockIdx.x, tm = blockIdx.y;

    int mrow = tm * 16 + l16;
    if (mrow >= M) mrow = M - 1;           // clamp for padded M (proj: M=4)
    const int ncol = tn * 16 + l16;        // N always multiple of 16 here

    v8f acc = {};
#pragma unroll 2
    for (int k0 = 0; k0 < K; k0 += 32) {
        // A fragment: lane half 0 -> K 0..15, half 1 -> K 16..31 (contiguous 16 per lane)
        const f4v* ap = (const f4v*)(A + (size_t)mrow * K + k0 + half * 16);
        // B fragment (W row-major (N,K), B[k][n] = W[n][k]):
        // lanes 0-15 K=0..15, lanes 16-31 K=16..31 (contiguous 16 per lane)
        const f4v* bp = (const f4v*)(W + (size_t)ncol * K + k0 + half * 16);

        float af[16], bf[16];
        *(f4v*)&af[0]  = ap[0]; *(f4v*)&af[4]  = ap[1];
        *(f4v*)&af[8]  = ap[2]; *(f4v*)&af[12] = ap[3];
        *(f4v*)&bf[0]  = bp[0]; *(f4v*)&bf[4]  = bp[1];
        *(f4v*)&bf[8]  = bp[2]; *(f4v*)&bf[12] = bp[3];

        v16h a, b;
#pragma unroll
        for (int i = 0; i < 16; ++i) {
            a[i] = (_Float16)af[i];
            b[i] = (_Float16)bf[i];
        }
        acc = __builtin_amdgcn_wmma_f32_16x16x32_f16(false, a, false, b,
                                                     (short)0, acc, false, false);
    }

    // Epilogue. C layout: VGPR r -> row = half*8 + r, col = lane&15.
    const float bv = bias[ncol];
    const int   m0 = tm * 16 + half * 8;
    float*       cp = C + (size_t)m0 * N + ncol;
    const float* rp = res + (size_t)m0 * N + ncol;  // unused unless RES

    if (tm * 16 + 16 <= M) {
        // full tile: uniform, no per-row exec masking
#pragma unroll
        for (int r = 0; r < 8; ++r) {
            float v = acc[r] + bv;
            if (RES) v += *rp;
            if (ACT == 1) v = fmaxf(v, 0.0f);
            if (ACT == 2) v = gelu_exact(v);
            *cp = v;
            cp += N;
            if (RES) rp += N;
        }
    } else {
        // partial tile (proj GEMM, M=4)
#pragma unroll
        for (int r = 0; r < 8; ++r) {
            if (m0 + r < M) {
                float v = acc[r] + bv;
                if (RES) v += *rp;
                if (ACT == 1) v = fmaxf(v, 0.0f);
                if (ACT == 2) v = gelu_exact(v);
                *cp = v;
            }
            cp += N;
            if (RES) rp += N;
        }
    }
}

// ---------------------------------------------------------------------------
__global__ void k_embed(const int* __restrict__ ids, const float* __restrict__ tbl,
                        float* __restrict__ x) {
    int row = blockIdx.x, t = threadIdx.x;
    x[(size_t)row * 256 + t] = tbl[(size_t)ids[row] * 256 + t];
}

// ---------------------------------------------------------------------------
// Flash-style attention, one block per (b, h). 128 threads = one q row each.
// K/V staged through LDS in 64-row chunks, online softmax in registers.
// qkv layout: (B*S, 768) rows: [q(256) | k(256) | v(256)], head slice h*64.
// ---------------------------------------------------------------------------
__global__ void k_attn(const float* __restrict__ qkv, const int* __restrict__ mask,
                       float* __restrict__ o) {
    __shared__ float sK[64 * 64];
    __shared__ float sV[64 * 64];
    __shared__ float sM[64];
    const int bh = blockIdx.x;
    const int b = bh >> 2, h = bh & 3;
    const int t = threadIdx.x;  // q row within sequence

    float q[64];
    const float* qp = qkv + (size_t)(b * 128 + t) * 768 + h * 64;
#pragma unroll
    for (int d = 0; d < 64; ++d) q[d] = qp[d] * 0.125f;  // 1/sqrt(64)

    float m = -3.0e38f, l = 0.0f, accv[64];
#pragma unroll
    for (int d = 0; d < 64; ++d) accv[d] = 0.0f;

    for (int ch = 0; ch < 2; ++ch) {
        __syncthreads();
        for (int idx = t; idx < 64 * 64; idx += 128) {
            int r = idx >> 6, d = idx & 63;
            size_t row = (size_t)(b * 128 + ch * 64 + r) * 768 + h * 64;
            sK[idx] = qkv[row + 256 + d];
            sV[idx] = qkv[row + 512 + d];
        }
        if (t < 64) sM[t] = (mask[b * 128 + ch * 64 + t] == 0) ? 1.0f : 0.0f;
        __syncthreads();

        for (int j = 0; j < 64; ++j) {
            const float* kr = &sK[j * 64];
            float s = 0.0f;
#pragma unroll
            for (int d = 0; d < 64; ++d) s += q[d] * kr[d];
            if (sM[j] != 0.0f) s = NEGV;
            float mn  = fmaxf(m, s);
            float cor = __expf(m - mn);
            float p   = __expf(s - mn);
            l = l * cor + p;
            const float* vr = &sV[j * 64];
#pragma unroll
            for (int d = 0; d < 64; ++d) accv[d] = accv[d] * cor + p * vr[d];
            m = mn;
        }
    }
    float inv = 1.0f / l;
    float* op = o + (size_t)(b * 128 + t) * 256 + h * 64;
#pragma unroll
    for (int d = 0; d < 64; ++d) op[d] = accv[d] * inv;
}

// ---------------------------------------------------------------------------
__global__ void k_ln(float* __restrict__ x, const float* __restrict__ g,
                     const float* __restrict__ b) {
    __shared__ float red[256];
    const int row = blockIdx.x, t = threadIdx.x;
    float v = x[(size_t)row * 256 + t];
    red[t] = v;
    __syncthreads();
    for (int s = 128; s > 0; s >>= 1) { if (t < s) red[t] += red[t + s]; __syncthreads(); }
    float mean = red[0] * (1.0f / 256.0f);
    __syncthreads();
    float d = v - mean;
    red[t] = d * d;
    __syncthreads();
    for (int s = 128; s > 0; s >>= 1) { if (t < s) red[t] += red[t + s]; __syncthreads(); }
    float r = rsqrtf(red[0] * (1.0f / 256.0f) + 1e-5f);
    x[(size_t)row * 256 + t] = d * r * g[t] + b[t];
}

__global__ void k_pool(const float* __restrict__ x, const int* __restrict__ mask,
                       float* __restrict__ pooled) {
    int b = blockIdx.x, d = threadIdx.x;
    float s = 0.0f, cnt = 0.0f;
    for (int t = 0; t < 128; ++t) {
        float am = (float)mask[b * 128 + t];
        s += x[(size_t)(b * 128 + t) * 256 + d] * am;
        cnt += am;
    }
    pooled[b * 256 + d] = s / fmaxf(cnt, 1.0f);
}

// ---------------------------------------------------------------------------
// tile_A: out[b,i,r,c] = A_base[b, idx(i), r, c & 63] * sa[idx]
// One block per (b, i, r) output row of length in_d; 64-float pattern in LDS,
// each thread's float4 is phase-invariant (stride 1024 == 0 mod 64) -> pure
// non-temporal b128 store stream. mode 0: A_HID, mode 1: A_INT.
// ---------------------------------------------------------------------------
__global__ void k_tile_A(float* __restrict__ out, const float* __restrict__ Abase,
                         const float* __restrict__ scales, int Ni, int in_d, int mode) {
    __shared__ float lds[64];
    const int bid = blockIdx.x, t = threadIdx.x;
    const int b = bid / (Ni * 16);
    const int rr = bid % (Ni * 16);
    const int i = rr >> 4, r = rr & 15;
    const int idx = (mode == 0) ? (i / 6) * 7 + (i % 6) : i * 7 + 6;
    if (t < 64)
        lds[t] = Abase[((size_t)(b * 224 + idx) * 16 + r) * 64 + t] * scales[idx * 2];
    __syncthreads();
    f4v val = *(const f4v*)&lds[(4 * t) & 63];
    float* orow = out + ((size_t)(b * Ni + i) * 16 + r) * in_d;
    const int nd4 = in_d >> 2;
    for (int c4 = t; c4 < nd4; c4 += 256)
        __builtin_nontemporal_store(val, (f4v*)(orow + 4 * c4));
}

// ---------------------------------------------------------------------------
// tile_B: out[b,i,c,r] = B_base[b, idx(i), c & 63, r] * sb[idx]
// One block per (b, i): the scaled 64x16 (= 1024 float) pattern goes to LDS,
// then is replicated out_d/64 times as 4 KB non-temporal store bursts.
// mode 0: B_HID, mode 1: B_KV, mode 2: B_INT.
// ---------------------------------------------------------------------------
__global__ void k_tile_B(float* __restrict__ out, const float* __restrict__ Bbase,
                         const float* __restrict__ scales, int Ni, int out_d, int mode) {
    __shared__ float lds[1024];
    const int bid = blockIdx.x, t = threadIdx.x;
    const int b = bid / Ni, i = bid % Ni;
    int idx;
    if (mode == 0)      idx = (i / 3) * 7 + (i % 3) * 3;
    else if (mode == 1) idx = (i / 2) * 7 + 1 + (i % 2);
    else                idx = (i / 2) * 7 + 4 + (i % 2);
    const float sb = scales[idx * 2 + 1];
    const float* src = Bbase + (size_t)(b * 224 + idx) * 1024;
#pragma unroll
    for (int j = 0; j < 4; ++j) lds[t * 4 + j] = src[t * 4 + j] * sb;
    __syncthreads();
    f4v val = *(const f4v*)&lds[t * 4];
    float* obase = out + (size_t)(b * Ni + i) * out_d * 16;
    const int reps = out_d >> 6;
    for (int rep = 0; rep < reps; ++rep)
        __builtin_nontemporal_store(val, (f4v*)(obase + rep * 1024 + t * 4));
}

// ---------------------------------------------------------------------------
extern "C" void kernel_launch(void* const* d_in, const int* in_sizes, int n_in,
                              void* d_out, int out_size, void* d_ws, size_t ws_size,
                              hipStream_t stream) {
    const int*   ids    = (const int*)d_in[0];
    const int*   amask  = (const int*)d_in[1];
    const float* embed  = (const float*)d_in[2];
    const float* qkv_w  = (const float*)d_in[3];
    const float* qkv_b  = (const float*)d_in[4];
    const float* out_w  = (const float*)d_in[5];
    const float* out_b  = (const float*)d_in[6];
    const float* ln1_g  = (const float*)d_in[7];
    const float* ln1_b  = (const float*)d_in[8];
    const float* ff1_w  = (const float*)d_in[9];
    const float* ff1_b  = (const float*)d_in[10];
    const float* ff2_w  = (const float*)d_in[11];
    const float* ff2_b  = (const float*)d_in[12];
    const float* ln2_g  = (const float*)d_in[13];
    const float* ln2_b  = (const float*)d_in[14];
    const float* proj_w = (const float*)d_in[15];
    const float* proj_b = (const float*)d_in[16];
    const float* a1w    = (const float*)d_in[17];
    const float* a1b    = (const float*)d_in[18];
    const float* a2w    = (const float*)d_in[19];
    const float* a2b    = (const float*)d_in[20];
    const float* b1w    = (const float*)d_in[21];
    const float* b1b    = (const float*)d_in[22];
    const float* b2w    = (const float*)d_in[23];
    const float* b2b    = (const float*)d_in[24];
    const float* scales = (const float*)d_in[25];
    float* out = (float*)d_out;

    // workspace carve-up (floats); total ~17.4 MB
    float* w      = (float*)d_ws;
    float* xA     = w; w += 512 * 256;        // activations ping
    float* xB     = w; w += 512 * 256;        // activations pong
    float* qkvb   = w; w += 512 * 768;        // fused qkv
    float* attno  = w; w += 512 * 256;        // attention output
    float* ff1b   = w; w += 512 * 2048;       // FF hidden
    float* pooled = w; w += 4 * 256;
    float* pe     = w; w += 4 * 114688;       // proj output (896 x 512)
    float* hbuf   = w; w += 896 * 256;        // dec hidden (reused A then B)
    float* Abase  = w; w += 896 * 1024;
    float* Bbase  = w; w += 896 * 1024;

    // ---- encoder ----
    k_embed<<<512, 256, 0, stream>>>(ids, embed, xA);
    for (int l = 0; l < 3; ++l) {
        k_gemm<0, false><<<dim3(768 / 16, 512 / 16), 32, 0, stream>>>(
            xA, qkv_w + (size_t)l * 768 * 256, qkv_b + l * 768, nullptr, qkvb, 512, 768, 256);
        k_attn<<<16, 128, 0, stream>>>(qkvb, amask, attno);
        k_gemm<0, true><<<dim3(256 / 16, 512 / 16), 32, 0, stream>>>(
            attno, out_w + (size_t)l * 256 * 256, out_b + l * 256, xA, xB, 512, 256, 256);
        k_ln<<<512, 256, 0, stream>>>(xB, ln1_g + l * 256, ln1_b + l * 256);
        k_gemm<1, false><<<dim3(2048 / 16, 512 / 16), 32, 0, stream>>>(
            xB, ff1_w + (size_t)l * 2048 * 256, ff1_b + l * 2048, nullptr, ff1b, 512, 2048, 256);
        k_gemm<0, true><<<dim3(256 / 16, 512 / 16), 32, 0, stream>>>(
            ff1b, ff2_w + (size_t)l * 256 * 2048, ff2_b + l * 256, xB, xA, 512, 256, 2048);
        k_ln<<<512, 256, 0, stream>>>(xA, ln2_g + l * 256, ln2_b + l * 256);
    }

    // ---- pool + big projection (reads proj_w exactly once) ----
    k_pool<<<4, 256, 0, stream>>>(xA, amask, pooled);
    k_gemm<0, false><<<dim3(114688 / 16, 1), 32, 0, stream>>>(
        pooled, proj_w, proj_b, nullptr, pe, 4, 114688, 256);

    // ---- decoders: pe viewed as (896, 512) ----
    k_gemm<2, false><<<dim3(256 / 16, 896 / 16), 32, 0, stream>>>(
        pe, a1w, a1b, nullptr, hbuf, 896, 256, 512);
    k_gemm<0, false><<<dim3(1024 / 16, 896 / 16), 32, 0, stream>>>(
        hbuf, a2w, a2b, nullptr, Abase, 896, 1024, 256);
    k_gemm<2, false><<<dim3(256 / 16, 896 / 16), 32, 0, stream>>>(
        pe, b1w, b1b, nullptr, hbuf, 896, 256, 512);
    k_gemm<0, false><<<dim3(1024 / 16, 896 / 16), 32, 0, stream>>>(
        hbuf, b2w, b2b, nullptr, Bbase, 896, 1024, 256);

    // ---- bandwidth-dominant tile broadcasts (589 MB NT stores) ----
    k_tile_A<<<4 * 192 * 16, 256, 0, stream>>>(out,               Abase, scales, 192, 4096,  0);
    k_tile_A<<<4 * 32 * 16,  256, 0, stream>>>(out + 50331648ull, Abase, scales, 32,  11008, 1);
    k_tile_B<<<4 * 96,       256, 0, stream>>>(out + 72876032ull, Bbase, scales, 96,  4096,  0);
    k_tile_B<<<4 * 64,       256, 0, stream>>>(out + 98041856ull, Bbase, scales, 64,  1024,  1);
    k_tile_B<<<4 * 64,       256, 0, stream>>>(out + 102236160ull, Bbase, scales, 64, 11008, 2);

    (void)in_sizes; (void)n_in; (void)out_size; (void)ws_size;
}
